// AttPool_59227599012342
// MI455X (gfx1250) — matrix operand order repaired and checked by reading, not compiled, not run on hardware
//
#include <hip/hip_runtime.h>
#include <hip/hip_bf16.h>
#include <math.h>

// Problem constants (match reference)
#define Bn     256
#define Nn     512
#define F_IN   128
#define Dd     512
#define OUTD   128
#define TNn    (Bn * Nn)        // 131072
#define En     (TNn * 16)       // 2097152

typedef __attribute__((ext_vector_type(16))) _Float16 v16h;
typedef __attribute__((ext_vector_type(8)))  float    v8f;
typedef __attribute__((ext_vector_type(4)))  int      v4i;

__device__ __forceinline__ v8f wmma_f16(v16h a, v16h b, v8f c) {
    return __builtin_amdgcn_wmma_f32_16x16x32_f16(false, a, false, b, (short)0, c, false, false);
}

// gfx1250 hardware V_TANH_F32 (TRANS32 class)
#if defined(__has_builtin)
#  if __has_builtin(__builtin_amdgcn_tanhf)
#    define TANH(x) __builtin_amdgcn_tanhf(x)
#  elif __has_builtin(__builtin_amdgcn_tanh_f32)
#    define TANH(x) __builtin_amdgcn_tanh_f32(x)
#  else
#    define TANH(x) tanhf(x)
#  endif
#else
#  define TANH(x) tanhf(x)
#endif

// gfx1250 async global->LDS copy (bypasses VGPRs, tracked by ASYNCcnt)
#if defined(__has_builtin)
#  if __has_builtin(__builtin_amdgcn_global_load_async_to_lds_b128)
#    define HAVE_ASYNC_LDS 1
#  endif
#endif

// Load 16 f16 down a column (row stride ld), K rows 0..7 then 16..23
__device__ __forceinline__ v16h load_b16(const _Float16* __restrict__ p, int ld) {
    v16h b;
#pragma unroll
    for (int j = 0; j < 16; ++j) b[j] = p[(size_t)((j & 7) + ((j >> 3) << 4)) * ld];
    return b;
}

// ---------------------------------------------------------------------------
// One-time f32 -> f16 weight conversion into workspace
// ---------------------------------------------------------------------------
__global__ void k_f32_to_f16(const float* __restrict__ s, _Float16* __restrict__ d, int n) {
    int i = blockIdx.x * 256 + threadIdx.x;
    if (i < n) d[i] = (_Float16)s[i];
}

// ---------------------------------------------------------------------------
// degs[i] = 1.0 + (#edges with dst==i)
// ---------------------------------------------------------------------------
__global__ void k_init_degs(float* __restrict__ degs) {
    int i = blockIdx.x * 256 + threadIdx.x;
    if (i < TNn) degs[i] = 1.0f;
}

__global__ void k_degree(const int* __restrict__ dst, float* __restrict__ degs) {
    int e = blockIdx.x * 256 + threadIdx.x;
    if (e < En) unsafeAtomicAdd(&degs[dst[e]], 1.0f);
}

// ---------------------------------------------------------------------------
// agg = h (self term).  h row stride is `hstride` (128 for input, 512 for cat)
// ---------------------------------------------------------------------------
__global__ void k_init_agg(const float* __restrict__ h, int hstride,
                           float* __restrict__ agg) {
    int t = blockIdx.x * 256 + threadIdx.x;     // TN*32 threads, 4 floats each
    int node = t >> 5, fg = t & 31;
    const float4 v = *(const float4*)(h + (size_t)node * hstride + fg * 4);
    *(float4*)(agg + (size_t)node * 128 + fg * 4) = v;
}

// agg[dst] += h[src] over all edges (float atomics resolve in L2; agg is L2-resident)
__global__ void k_scatter(const int* __restrict__ src, const int* __restrict__ dst,
                          const float* __restrict__ h, int hstride,
                          float* __restrict__ agg) {
    int t = blockIdx.x * 256 + threadIdx.x;     // E*32 threads
    int e = t >> 5, fg = t & 31;                // one wave per edge, 4 floats/lane
    int s = src[e], d = dst[e];
    const float4 v = *(const float4*)(h + (size_t)s * hstride + fg * 4);
    float* ap = agg + (size_t)d * 128 + fg * 4;
    unsafeAtomicAdd(ap + 0, v.x);
    unsafeAtomicAdd(ap + 1, v.y);
    unsafeAtomicAdd(ap + 2, v.z);
    unsafeAtomicAdd(ap + 3, v.w);
}

// ---------------------------------------------------------------------------
// cat[:, colBase:colBase+128] = tanh((agg @ W + b) / degs)
// Block = 256 threads (8 waves) = 128 rows x 128 cols.  W pre-converted f16,
// staged into LDS via async direct-to-LDS copies (padded stride for bank
// spread).  Each wave: 16 rows; column tiles processed in two halves of 4 so
// only 4 accumulators are live (avoids VGPR spills); B fragments prefetched
// one tile ahead of the WMMA consuming them.
// ---------------------------------------------------------------------------
#define SW_LD 136   // padded f16 row stride for W in LDS

__global__ void k_conv_gemm_tanh(const float* __restrict__ agg,
                                 const _Float16* __restrict__ W16,  // 128x128 f16
                                 const float* __restrict__ bias,    // 128
                                 const float* __restrict__ degs,
                                 float* __restrict__ cat,           // TN x 512
                                 int colBase) {
    __shared__ _Float16 sW[128 * SW_LD];
    __shared__ float    sBias[128];
    int tid = threadIdx.x;

#ifdef HAVE_ASYNC_LDS
    // 2048 16-byte chunks: 128 rows x 16 chunks; per-lane global & LDS addresses
    for (int c = tid; c < 2048; c += 256) {
        int r = c >> 4, cc = c & 15;
        __builtin_amdgcn_global_load_async_to_lds_b128(
            (__attribute__((address_space(1))) v4i*)(void*)(W16 + r * 128 + cc * 8),
            (__attribute__((address_space(3))) v4i*)(void*)&sW[r * SW_LD + cc * 8],
            0, 0);
    }
#else
    for (int i = tid; i < 128 * 128; i += 256)
        sW[(i >> 7) * SW_LD + (i & 127)] = W16[i];
#endif
    if (tid < 128) sBias[tid] = bias[tid];

    int wave = tid >> 5;
    int lane = tid & 31;
    int row0 = blockIdx.x * 128 + wave * 16;
    int m    = lane & 15;
    int koff = (lane & 16) ? 8 : 0;   // CDNA5 f16 A-layout: hi lanes hold K+8 / K+24
    int rbase = row0 + ((lane & 16) ? 8 : 0);

    // Preload A fragments (row m, all 128 K) and reciprocal degrees while the
    // async W staging is in flight.
    v16h afrag[4];
    const float* arow = agg + (size_t)(row0 + m) * 128;
#pragma unroll
    for (int kc = 0; kc < 4; ++kc) {
        const float* p = arow + kc * 32 + koff;
#pragma unroll
        for (int j = 0; j < 8; ++j) {
            afrag[kc][j]     = (_Float16)p[j];        // K = kc*32 + koff + j
            afrag[kc][8 + j] = (_Float16)p[16 + j];   // K = kc*32 + koff + 16 + j
        }
    }
    float dinv[8];
#pragma unroll
    for (int r = 0; r < 8; ++r) dinv[r] = 1.0f / degs[rbase + r];

#ifdef HAVE_ASYNC_LDS
    asm volatile("s_wait_asynccnt 0x0" ::: "memory");  // my async copies done
#endif
    __syncthreads();                                   // all waves' staging done

#pragma unroll
    for (int half = 0; half < 2; ++half) {
        v8f acc[4] = {};
#pragma unroll
        for (int kc = 0; kc < 4; ++kc) {
            const _Float16* brow = &sW[(kc * 32 + koff) * SW_LD + half * 64 + (lane & 15)];
            v16h bcur = load_b16(brow, SW_LD);
#pragma unroll
            for (int nt = 0; nt < 4; ++nt) {
                v16h bnext = bcur;
                if (nt < 3) bnext = load_b16(brow + (nt + 1) * 16, SW_LD); // prefetch
                acc[nt] = wmma_f16(afrag[kc], bcur, acc[nt]);
                bcur = bnext;
            }
        }
#pragma unroll
        for (int nt = 0; nt < 4; ++nt) {
            int colL = half * 64 + nt * 16 + (lane & 15);
            float bcol = sBias[colL];
#pragma unroll
            for (int r = 0; r < 8; ++r) {
                float lin = acc[nt][r] + bcol;
                cat[(size_t)(rbase + r) * Dd + colBase + colL] = TANH(lin * dinv[r]);
            }
        }
    }
}

// ---------------------------------------------------------------------------
// scores[row] = sum_d att_v[d] * tanh( (cat_row @ att_W)[d] + att_b[d] )
// Block = 256 threads (8 waves) = 16 rows x 512 cols; each wave owns 64 cols.
// A panel (16x512) staged as f16 in LDS (padded).  K=512 in 16 chunks of 32,
// double-buffered (A + 4 B fragments) so global B loads overlap the WMMAs.
// ---------------------------------------------------------------------------
#define SA_LD 520   // padded f16 row stride for A panel

__device__ __forceinline__ v16h load_afrag_lds(const _Float16* __restrict__ sA,
                                               int m, int kb, int koff) {
    const _Float16* ap = sA + m * SA_LD + kb + koff;
    v16h a;
#pragma unroll
    for (int j = 0; j < 8; ++j) { a[j] = ap[j]; a[8 + j] = ap[16 + j]; }
    return a;
}

__global__ void k_att_score(const float* __restrict__ cat,      // TN x 512
                            const _Float16* __restrict__ attW16,// 512 x 512 f16
                            const float* __restrict__ attB,     // 512
                            const float* __restrict__ attV,     // 512
                            float* __restrict__ scores) {       // TN
    __shared__ _Float16 sA[16 * SA_LD];
    __shared__ float    sScore[16];
    int tid  = threadIdx.x;
    int row0 = blockIdx.x * 16;

    for (int i = tid; i < 16 * 512; i += 256) {
        int r = i >> 9, c = i & 511;
        sA[r * SA_LD + c] = (_Float16)cat[(size_t)(row0 + r) * Dd + c];
    }
    if (tid < 16) sScore[tid] = 0.0f;
    __syncthreads();

    int wave = tid >> 5;
    int lane = tid & 31;
    int colW = wave * 64;
    int m    = lane & 15;
    int koff = (lane & 16) ? 8 : 0;

    v8f acc[4] = {};

    // Prologue: buffer "C" holds chunk 0
    v16h aC = load_afrag_lds(sA, m, 0, koff);
    const _Float16* bp0 = attW16 + (size_t)koff * Dd + colW + (lane & 15);
    v16h bC0 = load_b16(bp0, Dd);
    v16h bC1 = load_b16(bp0 + 16, Dd);
    v16h bC2 = load_b16(bp0 + 32, Dd);
    v16h bC3 = load_b16(bp0 + 48, Dd);

#pragma unroll 1
    for (int kc2 = 0; kc2 < 8; ++kc2) {
        int kbN = kc2 * 64 + 32;                 // odd chunk
        v16h aN = load_afrag_lds(sA, m, kbN, koff);
        const _Float16* bpN = attW16 + (size_t)(kbN + koff) * Dd + colW + (lane & 15);
        v16h bN0 = load_b16(bpN, Dd);
        v16h bN1 = load_b16(bpN + 16, Dd);
        v16h bN2 = load_b16(bpN + 32, Dd);
        v16h bN3 = load_b16(bpN + 48, Dd);

        acc[0] = wmma_f16(aC, bC0, acc[0]);
        acc[1] = wmma_f16(aC, bC1, acc[1]);
        acc[2] = wmma_f16(aC, bC2, acc[2]);
        acc[3] = wmma_f16(aC, bC3, acc[3]);

        if (kc2 < 7) {                           // prefetch next even chunk
            int kbC = kc2 * 64 + 64;
            aC = load_afrag_lds(sA, m, kbC, koff);
            const _Float16* bpC = attW16 + (size_t)(kbC + koff) * Dd + colW + (lane & 15);
            bC0 = load_b16(bpC, Dd);
            bC1 = load_b16(bpC + 16, Dd);
            bC2 = load_b16(bpC + 32, Dd);
            bC3 = load_b16(bpC + 48, Dd);
        }

        acc[0] = wmma_f16(aN, bN0, acc[0]);
        acc[1] = wmma_f16(aN, bN1, acc[1]);
        acc[2] = wmma_f16(aN, bN2, acc[2]);
        acc[3] = wmma_f16(aN, bN3, acc[3]);
    }

    int rloc = (lane & 16) ? 8 : 0;
#pragma unroll
    for (int nt = 0; nt < 4; ++nt) {
        int col = colW + nt * 16 + (lane & 15);
        float bb = attB[col], vv = attV[col];
#pragma unroll
        for (int r = 0; r < 8; ++r) {
            float y = acc[nt][r] + bb;
            atomicAdd(&sScore[rloc + r], vv * TANH(y));   // ds_add_f32 row reduce
        }
    }
    __syncthreads();
    if (tid < 16) scores[row0 + tid] = sScore[tid];
}

// ---------------------------------------------------------------------------
// Per-batch softmax over N=512 scores, then pooled[b,:] = sum_n attn[n]*cat[b,n,:]
// ---------------------------------------------------------------------------
__global__ void k_softmax_pool(const float* __restrict__ scores, // B*N
                               const float* __restrict__ cat,    // TN x 512
                               float* __restrict__ pooled) {     // B x 512
    __shared__ float sAttn[512];
    __shared__ float sRed[256];
    int b = blockIdx.x, tid = threadIdx.x;

    float s0 = scores[b * Nn + tid];
    float s1 = scores[b * Nn + 256 + tid];
    sRed[tid] = fmaxf(s0, s1);
    __syncthreads();
    for (int off = 128; off > 0; off >>= 1) {
        if (tid < off) sRed[tid] = fmaxf(sRed[tid], sRed[tid + off]);
        __syncthreads();
    }
    float gmax = sRed[0];
    __syncthreads();
    float e0 = __expf(s0 - gmax), e1 = __expf(s1 - gmax);
    sAttn[tid] = e0; sAttn[tid + 256] = e1;
    sRed[tid] = e0 + e1;
    __syncthreads();
    for (int off = 128; off > 0; off >>= 1) {
        if (tid < off) sRed[tid] += sRed[tid + off];
        __syncthreads();
    }
    float inv = 1.0f / sRed[0];
    __syncthreads();

    float p0 = 0.f, p1 = 0.f;
    int d0 = tid, d1 = tid + 256;
    const float* base = cat + (size_t)b * Nn * Dd;
#pragma unroll 4
    for (int n = 0; n < Nn; ++n) {
        float w = sAttn[n];
        p0 += w * base[(size_t)n * Dd + d0];
        p1 += w * base[(size_t)n * Dd + d1];
    }
    pooled[b * Dd + d0] = p0 * inv;
    pooled[b * Dd + d1] = p1 * inv;
}

// ---------------------------------------------------------------------------
// out = relu(pooled @ out_W + out_b)   (256x512)x(512x128): tiny, VALU is fine
// ---------------------------------------------------------------------------
__global__ void k_out(const float* __restrict__ pooled, // B x 512
                      const float* __restrict__ outW,   // 512 x 128
                      const float* __restrict__ outB,   // 128
                      float* __restrict__ out) {        // B x 128
    int idx = blockIdx.x * 256 + threadIdx.x;           // B*OUT
    int b = idx >> 7, o = idx & 127;
    const float* pr = pooled + (size_t)b * Dd;
    float acc = outB[o];
#pragma unroll 8
    for (int k = 0; k < Dd; ++k) acc += pr[k] * outW[k * OUTD + o];
    out[idx] = fmaxf(acc, 0.0f);
}

// ---------------------------------------------------------------------------
extern "C" void kernel_launch(void* const* d_in, const int* in_sizes, int n_in,
                              void* d_out, int out_size, void* d_ws, size_t ws_size,
                              hipStream_t stream) {
    const float* node_feat = (const float*)d_in[0];  // TN x 128
    const int*   edge_src  = (const int*)  d_in[1];  // E
    const int*   edge_dst  = (const int*)  d_in[2];  // E
    const float* conv_W    = (const float*)d_in[3];  // 4 x 128 x 128
    const float* conv_b    = (const float*)d_in[4];  // 4 x 128
    const float* att_W     = (const float*)d_in[5];  // 512 x 512
    const float* att_b     = (const float*)d_in[6];  // 512
    const float* att_v     = (const float*)d_in[7];  // 512
    const float* out_W     = (const float*)d_in[8];  // 512 x 128
    const float* out_b     = (const float*)d_in[9];  // 128
    float*       out       = (float*)d_out;          // B x 128

    float* ws = (float*)d_ws;
    float*     degs    = ws;                             // TN floats
    float*     scores  = ws + (size_t)TNn;               // TN floats
    float*     pooled  = ws + (size_t)2 * TNn;           // B*512 == TN floats
    _Float16*  attW16  = (_Float16*)(ws + (size_t)3 * TNn);  // 512*512 f16 (TN floats reserved)
    _Float16*  convW16 = (_Float16*)(ws + (size_t)4 * TNn);  // 4*128*128 f16
    float*     agg     = ws + (size_t)5 * TNn;           // TN*128 floats
    float*     cat     = ws + (size_t)5 * TNn + (size_t)TNn * 128; // TN*512 floats

    // one-time weight conversion to f16
    k_f32_to_f16<<<(Dd * Dd) / 256, 256, 0, stream>>>(att_W, attW16, Dd * Dd);
    k_f32_to_f16<<<(4 * 128 * 128) / 256, 256, 0, stream>>>(conv_W, convW16, 4 * 128 * 128);

    // degrees
    k_init_degs<<<TNn / 256, 256, 0, stream>>>(degs);
    k_degree<<<En / 256, 256, 0, stream>>>(edge_dst, degs);

    // 4 conv layers; layer outputs written directly into cat column blocks
    for (int l = 0; l < 4; ++l) {
        const float* hprev = (l == 0) ? node_feat : (cat + (size_t)(l - 1) * 128);
        int hstride        = (l == 0) ? F_IN : Dd;
        k_init_agg<<<(TNn * 32) / 256, 256, 0, stream>>>(hprev, hstride, agg);
        k_scatter<<<(En * 32) / 256, 256, 0, stream>>>(edge_src, edge_dst, hprev, hstride, agg);
        k_conv_gemm_tanh<<<TNn / 128, 256, 0, stream>>>(
            agg, convW16 + (size_t)l * 128 * 128, conv_b + (size_t)l * 128,
            degs, cat, l * 128);
    }

    // attention scores, softmax-pool, output layer
    k_att_score<<<TNn / 16, 256, 0, stream>>>(cat, attW16, att_b, att_v, scores);
    k_softmax_pool<<<Bn, 256, 0, stream>>>(scores, cat, pooled);
    k_out<<<(Bn * OUTD) / 256, 256, 0, stream>>>(pooled, out_W, out_b, out);
}
